// DN_21758304321882
// MI455X (gfx1250) — matrix operand order repaired and checked by reading, not compiled, not run on hardware
//
#include <hip/hip_runtime.h>

#define B_ROWS 256
#define HWK    1024
#define YDIM   65536
#define ZDIM   100

#define TILE_N 64            // y-columns per block (4 n-tiles of 16)
#define NT     4
#define KC     64            // K chunk per async stage
#define NCHUNK (HWK / KC)    // 16
#define KCP    68            // staging f32 row stride (bank stride 4, conflict-free b128)
#define FSTR   72            // f16 tile row stride in halfs (144B, bank stride 36, conflict-free)

typedef __attribute__((ext_vector_type(16))) _Float16 v16h;
typedef __attribute__((ext_vector_type(8)))  float    v8f;

union Frag { uint4 u[2]; v16h h; };
union H8   { uint4 u; _Float16 s[8]; };

typedef __attribute__((address_space(3))) void lds_void;

// 32-bit LDS byte address of a generic pointer that points into LDS
__device__ __forceinline__ unsigned lds_addr(const void* p) {
    return (unsigned)(unsigned long long)(lds_void*)p;
}

// CDNA5 async HBM->LDS copy, 16B per lane, tracked on ASYNCcnt
__device__ __forceinline__ void async_copy_b128(unsigned lds_off, const float* gaddr) {
    asm volatile("global_load_async_to_lds_b128 %0, %1, off"
                 :: "v"(lds_off), "v"(gaddr) : "memory");
}
__device__ __forceinline__ void wait_async0() {
    asm volatile("s_wait_asynccnt 0" ::: "memory");
}

// monotonic unsigned encoding of float (handles negatives), bigger == larger float
__device__ __forceinline__ unsigned ordf(float f) {
    unsigned u = __float_as_uint(f);
    return (u & 0x80000000u) ? ~u : (u | 0x80000000u);
}

// ---------------------------------------------------------------------------
// Kernel 1: xf = l2norm(x rows) -> f16, and init packed-argmax slots
// ---------------------------------------------------------------------------
__global__ void __launch_bounds__(256)
prep_kernel(const float* __restrict__ x, _Float16* __restrict__ xf16,
            unsigned long long* __restrict__ packed) {
    __shared__ float red[256];
    const int b = blockIdx.x, t = threadIdx.x;
    const float4 v = *(const float4*)(x + (size_t)b * HWK + t * 4);
    red[t] = v.x * v.x + v.y * v.y + v.z * v.z + v.w * v.w;
    __syncthreads();
    for (int s = 128; s > 0; s >>= 1) {
        if (t < s) red[t] += red[t + s];
        __syncthreads();
    }
    const float inv = 1.0f / fmaxf(sqrtf(red[0]), 1e-12f);
    _Float16* o = xf16 + (size_t)b * HWK + t * 4;
    o[0] = (_Float16)(v.x * inv);
    o[1] = (_Float16)(v.y * inv);
    o[2] = (_Float16)(v.z * inv);
    o[3] = (_Float16)(v.w * inv);
    if (b == 0) packed[t] = 0ull;
}

// ---------------------------------------------------------------------------
// Kernel 2: fused (xf @ norm(w)^T) * (age>=1) -> per-b packed argmax over y
// Async staging (f32, double buffered) -> convert pass (f16 tile) -> WMMA.
// ---------------------------------------------------------------------------
__global__ void __launch_bounds__(256)
gemm_argmax_kernel(const _Float16* __restrict__ xf16,
                   const float* __restrict__ x2y_w,
                   const float* __restrict__ age,
                   unsigned long long* __restrict__ packed) {
    __shared__ float    sbuf[2][TILE_N][KCP];   // 34,816 B async staging (f32)
    __shared__ _Float16 wtile[TILE_N][FSTR];    //  9,216 B fragment tile (f16)
    __shared__ float    ssum[TILE_N];
    __shared__ float    scale_sh[TILE_N];

    const int tid = threadIdx.x;
    const int yb  = blockIdx.x * TILE_N;

    if (tid < TILE_N) ssum[tid] = 0.0f;

    const int cr  = tid >> 2;          // copy/convert row 0..63
    const int cc0 = (tid & 3) * 16;    // copy/convert col base (elements)

    auto issue_chunk = [&](int chunk, int bb) {
        const float* g = x2y_w + (size_t)(yb + cr) * HWK + chunk * KC + cc0;
        const unsigned l = lds_addr(&sbuf[bb][cr][cc0]);
#pragma unroll
        for (int i = 0; i < 4; ++i)
            async_copy_b128(l + i * 16, g + i * 4);
    };

    const int lane  = tid & 31;
    const int wv    = tid >> 5;
    const int Mb0   = wv * 32;
    const int Mb1   = Mb0 + 16;
    const int lrow  = lane & 15;
    const int lhalf = lane >> 4;

    v8f acc[2][NT] = {};

    // A fragment (16-bit 16x32 layout): two 16B chunks at K=lhalf*16(+8)
    const _Float16* a0base = xf16 + (size_t)(Mb0 + lrow) * HWK + lhalf * 16;
    const _Float16* a1base = xf16 + (size_t)(Mb1 + lrow) * HWK + lhalf * 16;
    // B fragment (32x16 layout): N=lane&15 -> tile row, 16 consecutive halfs
    const _Float16* bbase = &wtile[lrow][lhalf * 16];

    float sspart = 0.0f;
    int buf = 0;
    issue_chunk(0, 0);

    for (int c = 0; c < NCHUNK; ++c) {
        wait_async0();
        __syncthreads();                        // staging[buf] visible to all
        if (c + 1 < NCHUNK) issue_chunk(c + 1, buf ^ 1);

        // ---- convert pass: staging f32 -> f16 tile, fused ssum partial ----
        {
            const float4* p = (const float4*)&sbuf[buf][cr][cc0];
            const float4 f0 = p[0], f1 = p[1], f2 = p[2], f3 = p[3];
            sspart += f0.x * f0.x + f0.y * f0.y + f0.z * f0.z + f0.w * f0.w
                    + f1.x * f1.x + f1.y * f1.y + f1.z * f1.z + f1.w * f1.w
                    + f2.x * f2.x + f2.y * f2.y + f2.z * f2.z + f2.w * f2.w
                    + f3.x * f3.x + f3.y * f3.y + f3.z * f3.z + f3.w * f3.w;
            H8 h0, h1;
            h0.s[0] = (_Float16)f0.x; h0.s[1] = (_Float16)f0.y;
            h0.s[2] = (_Float16)f0.z; h0.s[3] = (_Float16)f0.w;
            h0.s[4] = (_Float16)f1.x; h0.s[5] = (_Float16)f1.y;
            h0.s[6] = (_Float16)f1.z; h0.s[7] = (_Float16)f1.w;
            h1.s[0] = (_Float16)f2.x; h1.s[1] = (_Float16)f2.y;
            h1.s[2] = (_Float16)f2.z; h1.s[3] = (_Float16)f2.w;
            h1.s[4] = (_Float16)f3.x; h1.s[5] = (_Float16)f3.y;
            h1.s[6] = (_Float16)f3.z; h1.s[7] = (_Float16)f3.w;
            *(uint4*)&wtile[cr][cc0]     = h0.u;
            *(uint4*)&wtile[cr][cc0 + 8] = h1.u;
        }
        __syncthreads();                        // f16 tile ready

        // ---- compute: all loads issued up front, then 8 WMMAs ----
#pragma unroll
        for (int ks = 0; ks < KC / 32; ++ks) {
            const int kg = c * KC + ks * 32;    // global K for A
            Frag fa0, fa1, fb0, fb1, fb2, fb3;
            fa0.u[0] = *(const uint4*)(a0base + kg);
            fa0.u[1] = *(const uint4*)(a0base + kg + 8);
            fa1.u[0] = *(const uint4*)(a1base + kg);
            fa1.u[1] = *(const uint4*)(a1base + kg + 8);
            const _Float16* bp = bbase + ks * 32;
            fb0.u[0] = *(const uint4*)(bp);
            fb0.u[1] = *(const uint4*)(bp + 8);
            fb1.u[0] = *(const uint4*)(bp + 16 * FSTR);
            fb1.u[1] = *(const uint4*)(bp + 16 * FSTR + 8);
            fb2.u[0] = *(const uint4*)(bp + 32 * FSTR);
            fb2.u[1] = *(const uint4*)(bp + 32 * FSTR + 8);
            fb3.u[0] = *(const uint4*)(bp + 48 * FSTR);
            fb3.u[1] = *(const uint4*)(bp + 48 * FSTR + 8);

            acc[0][0] = __builtin_amdgcn_wmma_f32_16x16x32_f16(
                false, fa0.h, false, fb0.h, (short)0, acc[0][0], false, false);
            acc[1][0] = __builtin_amdgcn_wmma_f32_16x16x32_f16(
                false, fa1.h, false, fb0.h, (short)0, acc[1][0], false, false);
            acc[0][1] = __builtin_amdgcn_wmma_f32_16x16x32_f16(
                false, fa0.h, false, fb1.h, (short)0, acc[0][1], false, false);
            acc[1][1] = __builtin_amdgcn_wmma_f32_16x16x32_f16(
                false, fa1.h, false, fb1.h, (short)0, acc[1][1], false, false);
            acc[0][2] = __builtin_amdgcn_wmma_f32_16x16x32_f16(
                false, fa0.h, false, fb2.h, (short)0, acc[0][2], false, false);
            acc[1][2] = __builtin_amdgcn_wmma_f32_16x16x32_f16(
                false, fa1.h, false, fb2.h, (short)0, acc[1][2], false, false);
            acc[0][3] = __builtin_amdgcn_wmma_f32_16x16x32_f16(
                false, fa0.h, false, fb3.h, (short)0, acc[0][3], false, false);
            acc[1][3] = __builtin_amdgcn_wmma_f32_16x16x32_f16(
                false, fa1.h, false, fb3.h, (short)0, acc[1][3], false, false);
        }
        __syncthreads();                        // reads of wtile/staging[buf] done
        buf ^= 1;
    }

    atomicAdd(&ssum[cr], sspart);               // ds_add_f32
    __syncthreads();
    if (tid < TILE_N) {
        const float inv = 1.0f / fmaxf(sqrtf(ssum[tid]), 1e-12f);
        const float act = (age[yb + tid] >= 1.0f) ? 1.0f : 0.0f;
        scale_sh[tid] = inv * act;
    }
    __syncthreads();

    // Epilogue: scale, local max over 4 n-tiles, shuffle-reduce 16 N-lanes,
    // one packed u64 atomicMax per batch row per block.
#pragma unroll
    for (int j = 0; j < 8; ++j) {
#pragma unroll
        for (int m = 0; m < 2; ++m) {
            unsigned long long key = 0ull;
#pragma unroll
            for (int nt = 0; nt < NT; ++nt) {
                const float v = acc[m][nt][j] * scale_sh[nt * 16 + lrow];
                const unsigned yg = (unsigned)(yb + nt * 16 + lrow);
                const unsigned long long k2 =
                    ((unsigned long long)ordf(v) << 32) | (0xFFFFFFFFu - yg);
                if (k2 > key) key = k2;
            }
#pragma unroll
            for (int off = 8; off >= 1; off >>= 1) {
                unsigned hi = (unsigned)(key >> 32);
                unsigned lo = (unsigned)key;
                const unsigned ohi = __shfl_xor(hi, off, 16);
                const unsigned olo = __shfl_xor(lo, off, 16);
                const unsigned long long ok =
                    ((unsigned long long)ohi << 32) | (unsigned long long)olo;
                if (ok > key) key = ok;
            }
            if (lrow == 0) {
                const int brow = (m == 0 ? Mb0 : Mb1) + lhalf * 8 + j;
                atomicMax(&packed[brow], key);
            }
        }
    }
}

// ---------------------------------------------------------------------------
// Kernel 3: decode argmax y; output[b,:] = y2z_w[:, y]; z_pre[b] = argmax_z
// ---------------------------------------------------------------------------
__global__ void __launch_bounds__(128)
finalize_kernel(const unsigned long long* __restrict__ packed,
                const float* __restrict__ y2z_w,
                float* __restrict__ out) {
    __shared__ float sv[128];
    __shared__ int   si[128];
    const int b = blockIdx.x, t = threadIdx.x;
    const unsigned y = 0xFFFFFFFFu - (unsigned)(packed[b] & 0xFFFFFFFFull);

    float v = -3.0e38f;
    if (t < ZDIM) {
        v = y2z_w[(size_t)t * YDIM + y];
        out[(size_t)b * ZDIM + t] = v;
    }
    sv[t] = v;
    si[t] = t;
    __syncthreads();
    for (int s = 64; s > 0; s >>= 1) {
        if (t < s) {
            const float ov = sv[t + s];
            const int   oi = si[t + s];
            if (ov > sv[t] || (ov == sv[t] && oi < si[t])) { sv[t] = ov; si[t] = oi; }
        }
        __syncthreads();
    }
    if (t == 0) out[(size_t)B_ROWS * ZDIM + b] = (float)si[0];
}

// ---------------------------------------------------------------------------
extern "C" void kernel_launch(void* const* d_in, const int* in_sizes, int n_in,
                              void* d_out, int out_size, void* d_ws, size_t ws_size,
                              hipStream_t stream) {
    const float* x      = (const float*)d_in[0];   // (256,32,32) f32
    // d_in[1] = z (unused by outputs)
    const float* x2y_w  = (const float*)d_in[2];   // (65536,1024) f32
    const float* y2z_w  = (const float*)d_in[3];   // (100,65536)  f32
    const float* age    = (const float*)d_in[4];   // (1,65536)    f32
    // d_in[5] = test_cnt (0 -> top-1/argmax)

    _Float16* xf16 = (_Float16*)d_ws;                                   // 512 KB
    unsigned long long* packed =
        (unsigned long long*)((char*)d_ws + (size_t)B_ROWS * HWK * 2);  // 2 KB
    float* out = (float*)d_out;   // 256*100 output, then 256 z_pre (as float)

    prep_kernel<<<B_ROWS, 256, 0, stream>>>(x, xf16, packed);
    gemm_argmax_kernel<<<YDIM / TILE_N, 256, 0, stream>>>(xf16, x2y_w, age, packed);
    finalize_kernel<<<B_ROWS, 128, 0, stream>>>(packed, y2z_w, out);
}
